// NonLinearMessagePassingLayer_32109175505235
// MI455X (gfx1250) — matrix operand order repaired and checked by reading, not compiled, not run on hardware
//
#include <hip/hip_runtime.h>
#include <hip/hip_bf16.h>

// ---------------------------------------------------------------------------
// Types for CDNA5 WMMA
// ---------------------------------------------------------------------------
typedef __attribute__((ext_vector_type(8)))  float  v8f;
typedef __attribute__((ext_vector_type(16))) __bf16 v16bf;
typedef __attribute__((ext_vector_type(8)))  __bf16 v8bf;

union BF16x16 { v16bf v; v8bf h[2]; };

#define DD   128          // feature dim
#define HH   256          // hidden dim
#define K1E  384          // 3*D message-in dim
#define ET   128          // edge/node rows per block
#define ASTR 392          // LDS A stride (384 + 8 pad) for edge kernels
#define NSTR 264          // LDS stride (256 + 8 pad) for hidden / node-A
#define LN_EPS 1e-6f

static __device__ inline unsigned short f2bf(float f) {
    unsigned u = __float_as_uint(f);
    unsigned r = (u + 0x7FFFu + ((u >> 16) & 1u)) >> 16;
    return (unsigned short)r;
}

// A-fragment (16x32 bf16) from row-major LDS tile.
// lane<16: rows 0..15, K {kb+0..7, kb+16..23}; lane>=16: K {kb+8..15, kb+24..31}
static __device__ inline v16bf load_a_frag(const unsigned short* base, int stride,
                                           int lane, int kb) {
    int half = lane >> 4;
    int row  = lane & 15;
    const unsigned short* p = base + row * stride + kb + half * 8;
    BF16x16 f;
    f.h[0] = *reinterpret_cast<const v8bf*>(p);
    f.h[1] = *reinterpret_cast<const v8bf*>(p + 16);
    return f.v;
}

// B-fragment (32x16 bf16) from global weights stored transposed as [N][K].
// lane<16: K {kb+0..7, kb+16..23}; lane>=16: K {kb+8..15, kb+24..31}; column = nb + (lane&15)
static __device__ inline v16bf load_b_frag(const unsigned short* wt, int K,
                                           int lane, int nb, int kb) {
    int col = nb + (lane & 15);
    int kh  = kb + ((lane >> 4) << 3);
    const unsigned short* p = wt + (long)col * K + kh;
    BF16x16 f;
    f.h[0] = *reinterpret_cast<const v8bf*>(p);
    f.h[1] = *reinterpret_cast<const v8bf*>(p + 16);
    return f.v;
}

static __device__ inline v8f wmma_bf16(v16bf a, v16bf b, v8f c) {
    return __builtin_amdgcn_wmma_f32_16x16x32_bf16(false, a, false, b,
                                                   (short)0, c, false, false);
}

// ---------------------------------------------------------------------------
// Prep kernels
// ---------------------------------------------------------------------------
__global__ void transpose_to_bf16(const float* __restrict__ w,
                                  unsigned short* __restrict__ wt,
                                  int K, int N) {
    long idx = (long)blockIdx.x * blockDim.x + threadIdx.x;
    if (idx < (long)K * N) {
        int k = (int)(idx / N), n = (int)(idx % N);
        wt[(long)n * K + k] = f2bf(w[idx]);
    }
}

__global__ void zero_f32(float* __restrict__ p, long n) {
    long idx = (long)blockIdx.x * blockDim.x + threadIdx.x;
    if (idx < n) p[idx] = 0.f;
}

// ---------------------------------------------------------------------------
// Edge-side kernel. MODE 0: message MLP -> atomic segment-sum into agg.
//                   MODE 1: edge MLP + edges@We residual + LayerNorm -> out.
// Block = 256 threads (8 waves). Each block handles 128 edges; each wave
// owns one 16-row M-tile.
// ---------------------------------------------------------------------------
template <int MODE>
__global__ __launch_bounds__(256) void edge_mlp_kernel(
    const float* __restrict__ nodes, const float* __restrict__ edges,
    const int* __restrict__ senders, const int* __restrict__ receivers,
    const unsigned short* __restrict__ w1t, const float* __restrict__ b1,
    const unsigned short* __restrict__ w2t, const float* __restrict__ b2,
    const unsigned short* __restrict__ Wrt,                 // residual proj (MODE 1)
    const float* __restrict__ ln_s, const float* __restrict__ ln_b,
    float* __restrict__ agg, float* __restrict__ out_edges, int E)
{
    extern __shared__ unsigned short sm[];
    unsigned short* As = sm;                 // [128][ASTR] bf16 message_in tile
    unsigned short* Hs = sm + ET * ASTR;     // [128][NSTR] bf16 hidden tile

    const int tid  = threadIdx.x;
    const int lane = tid & 31;
    const int wave = tid >> 5;
    const long e0  = (long)blockIdx.x * ET;

    // ---- build message_in tile: [sender | receiver | edge] -> bf16 LDS ----
    {
        int  r     = tid >> 1;              // row 0..127 (2 threads per row)
        int  cbase = (tid & 1) * 192;       // half of 384 columns
        long e     = e0 + r;
        bool valid = (e < E);
        int  s = 0, rc = 0;
        if (valid) { s = senders[e]; rc = receivers[e]; }
        unsigned short* dst = As + r * ASTR;
        for (int c = cbase; c < cbase + 192; c += 4) {
            float4 v = make_float4(0.f, 0.f, 0.f, 0.f);
            if (valid) {
                const float* src = (c < 128) ? (nodes + (long)s  * DD + c)
                                 : (c < 256) ? (nodes + (long)rc * DD + (c - 128))
                                             : (edges + e        * DD + (c - 256));
                v = *reinterpret_cast<const float4*>(src);
            }
            uint2 pk;
            pk.x = (unsigned)f2bf(v.x) | ((unsigned)f2bf(v.y) << 16);
            pk.y = (unsigned)f2bf(v.z) | ((unsigned)f2bf(v.w) << 16);
            *reinterpret_cast<uint2*>(dst + c) = pk;
        }
    }
    __syncthreads();

    const unsigned short* Abase = As + wave * 16 * ASTR;
    const int nlo = lane & 15, hi = lane >> 4;

    // ---- GEMM1: X[16,384] @ W1[384,256], relu, stash bf16 hidden in LDS ----
    for (int pass = 0; pass < 2; ++pass) {
        v8f acc[8] = {};
        for (int kt = 0; kt < 12; ++kt) {
            v16bf a = load_a_frag(Abase, ASTR, lane, kt * 32);
#pragma unroll
            for (int nt = 0; nt < 8; ++nt) {
                v16bf b = load_b_frag(w1t, K1E, lane, (pass * 8 + nt) * 16, kt * 32);
                acc[nt] = wmma_bf16(a, b, acc[nt]);
            }
        }
#pragma unroll
        for (int nt = 0; nt < 8; ++nt) {
            int   n    = (pass * 8 + nt) * 16 + nlo;
            float bias = b1[n];
#pragma unroll
            for (int r = 0; r < 8; ++r) {
                int row = wave * 16 + r + hi * 8;
                Hs[row * NSTR + n] = f2bf(fmaxf(acc[nt][r] + bias, 0.f));
            }
        }
    }
    __syncthreads();

    // ---- GEMM2: relu(h)[16,256] @ W2[256,128] ----
    v8f acc2[8] = {};
    const unsigned short* Hbase = Hs + wave * 16 * NSTR;
    for (int kt = 0; kt < 8; ++kt) {
        v16bf a = load_a_frag(Hbase, NSTR, lane, kt * 32);
#pragma unroll
        for (int nt = 0; nt < 8; ++nt) {
            v16bf b = load_b_frag(w2t, HH, lane, nt * 16, kt * 32);
            acc2[nt] = wmma_bf16(a, b, acc2[nt]);
        }
    }

    if (MODE == 1) {
        // residual: edges @ We — A = columns 256..383 of the message_in tile
        for (int kt = 0; kt < 4; ++kt) {
            v16bf a = load_a_frag(Abase, ASTR, lane, 256 + kt * 32);
#pragma unroll
            for (int nt = 0; nt < 8; ++nt) {
                v16bf b = load_b_frag(Wrt, DD, lane, nt * 16, kt * 32);
                acc2[nt] = wmma_bf16(a, b, acc2[nt]);
            }
        }
    }

    float bs[8];
#pragma unroll
    for (int nt = 0; nt < 8; ++nt) bs[nt] = b2[nt * 16 + nlo];

    if (MODE == 0) {
        // scatter-add messages into aggregated[receiver]
        for (int r = 0; r < 8; ++r) {
            long e = e0 + wave * 16 + r + hi * 8;
            if (e < E) {
                int   rc  = receivers[e];
                float* dp = agg + (long)rc * DD + nlo;
#pragma unroll
                for (int nt = 0; nt < 8; ++nt)
                    atomicAdd(dp + nt * 16, acc2[nt][r] + bs[nt]);
            }
        }
    } else {
        // bias + LayerNorm; per-wave fp32 scratch overlays this wave's H rows
        float* scr   = reinterpret_cast<float*>(Hs + wave * 16 * NSTR);
        float* stats = scr + 16 * 128;      // 32 floats fit in stride padding
#pragma unroll
        for (int nt = 0; nt < 8; ++nt)
#pragma unroll
            for (int r = 0; r < 8; ++r)
                scr[(r + hi * 8) * 128 + nt * 16 + nlo] = acc2[nt][r] + bs[nt];
        __syncthreads();
        if (lane < 16) {
            float s = 0.f, ss = 0.f;
            for (int c = 0; c < 128; ++c) {
                float v = scr[lane * 128 + c];
                s += v; ss += v * v;
            }
            float mu  = s  * (1.f / 128.f);
            float var = ss * (1.f / 128.f) - mu * mu;
            stats[lane * 2]     = mu;
            stats[lane * 2 + 1] = rsqrtf(var + LN_EPS);
        }
        __syncthreads();
        for (int t = 0; t < 64; ++t) {
            int  idx = t * 32 + lane;
            int  row = idx >> 7, col = idx & 127;
            long e   = e0 + wave * 16 + row;
            if (e < E) {
                float v = scr[row * 128 + col];
                out_edges[e * DD + col] =
                    (v - stats[row * 2]) * stats[row * 2 + 1] * ln_s[col] + ln_b[col];
            }
        }
    }
}

// ---------------------------------------------------------------------------
// Node kernel: [nodes | aggregated] -> MLP -> + nodes@Wn -> LayerNorm -> out
// ---------------------------------------------------------------------------
__global__ __launch_bounds__(256) void node_mlp_kernel(
    const float* __restrict__ nodes, const float* __restrict__ agg,
    const unsigned short* __restrict__ w1t, const float* __restrict__ b1,
    const unsigned short* __restrict__ w2t, const float* __restrict__ b2,
    const unsigned short* __restrict__ Wnt,
    const float* __restrict__ ln_s, const float* __restrict__ ln_b,
    float* __restrict__ out_nodes, int N)
{
    extern __shared__ unsigned short sm[];
    unsigned short* As = sm;                 // [128][NSTR] bf16 [node|agg]
    unsigned short* Hs = sm + ET * NSTR;     // [128][NSTR] bf16 hidden

    const int tid  = threadIdx.x;
    const int lane = tid & 31;
    const int wave = tid >> 5;
    const long i0  = (long)blockIdx.x * ET;

    {
        int  r     = tid >> 1;
        int  cbase = (tid & 1) * 128;
        long i     = i0 + r;
        bool valid = (i < N);
        unsigned short* dst = As + r * NSTR;
        for (int c = cbase; c < cbase + 128; c += 4) {
            float4 v = make_float4(0.f, 0.f, 0.f, 0.f);
            if (valid) {
                const float* src = (c < 128) ? (nodes + i * DD + c)
                                             : (agg   + i * DD + (c - 128));
                v = *reinterpret_cast<const float4*>(src);
            }
            uint2 pk;
            pk.x = (unsigned)f2bf(v.x) | ((unsigned)f2bf(v.y) << 16);
            pk.y = (unsigned)f2bf(v.z) | ((unsigned)f2bf(v.w) << 16);
            *reinterpret_cast<uint2*>(dst + c) = pk;
        }
    }
    __syncthreads();

    const unsigned short* Abase = As + wave * 16 * NSTR;
    const int nlo = lane & 15, hi = lane >> 4;

    for (int pass = 0; pass < 2; ++pass) {
        v8f acc[8] = {};
        for (int kt = 0; kt < 8; ++kt) {
            v16bf a = load_a_frag(Abase, NSTR, lane, kt * 32);
#pragma unroll
            for (int nt = 0; nt < 8; ++nt) {
                v16bf b = load_b_frag(w1t, 2 * DD, lane, (pass * 8 + nt) * 16, kt * 32);
                acc[nt] = wmma_bf16(a, b, acc[nt]);
            }
        }
#pragma unroll
        for (int nt = 0; nt < 8; ++nt) {
            int   n    = (pass * 8 + nt) * 16 + nlo;
            float bias = b1[n];
#pragma unroll
            for (int r = 0; r < 8; ++r) {
                int row = wave * 16 + r + hi * 8;
                Hs[row * NSTR + n] = f2bf(fmaxf(acc[nt][r] + bias, 0.f));
            }
        }
    }
    __syncthreads();

    v8f acc2[8] = {};
    const unsigned short* Hbase = Hs + wave * 16 * NSTR;
    for (int kt = 0; kt < 8; ++kt) {
        v16bf a = load_a_frag(Hbase, NSTR, lane, kt * 32);
#pragma unroll
        for (int nt = 0; nt < 8; ++nt) {
            v16bf b = load_b_frag(w2t, HH, lane, nt * 16, kt * 32);
            acc2[nt] = wmma_bf16(a, b, acc2[nt]);
        }
    }
    // residual nodes @ Wn : A = columns 0..127
    for (int kt = 0; kt < 4; ++kt) {
        v16bf a = load_a_frag(Abase, NSTR, lane, kt * 32);
#pragma unroll
        for (int nt = 0; nt < 8; ++nt) {
            v16bf b = load_b_frag(Wnt, DD, lane, nt * 16, kt * 32);
            acc2[nt] = wmma_bf16(a, b, acc2[nt]);
        }
    }

    float bs[8];
#pragma unroll
    for (int nt = 0; nt < 8; ++nt) bs[nt] = b2[nt * 16 + nlo];

    float* scr   = reinterpret_cast<float*>(Hs + wave * 16 * NSTR);
    float* stats = scr + 16 * 128;
#pragma unroll
    for (int nt = 0; nt < 8; ++nt)
#pragma unroll
        for (int r = 0; r < 8; ++r)
            scr[(r + hi * 8) * 128 + nt * 16 + nlo] = acc2[nt][r] + bs[nt];
    __syncthreads();
    if (lane < 16) {
        float s = 0.f, ss = 0.f;
        for (int c = 0; c < 128; ++c) {
            float v = scr[lane * 128 + c];
            s += v; ss += v * v;
        }
        float mu  = s  * (1.f / 128.f);
        float var = ss * (1.f / 128.f) - mu * mu;
        stats[lane * 2]     = mu;
        stats[lane * 2 + 1] = rsqrtf(var + LN_EPS);
    }
    __syncthreads();
    for (int t = 0; t < 64; ++t) {
        int  idx = t * 32 + lane;
        int  row = idx >> 7, col = idx & 127;
        long i   = i0 + wave * 16 + row;
        if (i < N) {
            float v = scr[row * 128 + col];
            out_nodes[i * DD + col] =
                (v - stats[row * 2]) * stats[row * 2 + 1] * ln_s[col] + ln_b[col];
        }
    }
}

// ---------------------------------------------------------------------------
// Host launcher
// ---------------------------------------------------------------------------
extern "C" void kernel_launch(void* const* d_in, const int* in_sizes, int n_in,
                              void* d_out, int out_size, void* d_ws, size_t ws_size,
                              hipStream_t stream) {
    const float* nodes     = (const float*)d_in[0];
    const float* edges     = (const float*)d_in[1];
    const int*   senders   = (const int*)  d_in[2];
    const int*   receivers = (const int*)  d_in[3];
    const float* msg_w1  = (const float*)d_in[4];
    const float* msg_b1  = (const float*)d_in[5];
    const float* msg_w2  = (const float*)d_in[6];
    const float* msg_b2  = (const float*)d_in[7];
    const float* node_w1 = (const float*)d_in[8];
    const float* node_b1 = (const float*)d_in[9];
    const float* node_w2 = (const float*)d_in[10];
    const float* node_b2 = (const float*)d_in[11];
    const float* edge_w1 = (const float*)d_in[12];
    const float* edge_b1 = (const float*)d_in[13];
    const float* edge_w2 = (const float*)d_in[14];
    const float* edge_b2 = (const float*)d_in[15];
    const float* Wn      = (const float*)d_in[16];
    const float* We      = (const float*)d_in[17];
    const float* ln_s    = (const float*)d_in[18];
    const float* ln_b    = (const float*)d_in[19];

    const int N = in_sizes[0] / DD;   // 50000
    const int E = in_sizes[1] / DD;   // 600000

    // workspace layout (bf16 transposed weights, then fp32 aggregated)
    char* ws = (char*)d_ws;
    unsigned short* msg_w1t  = (unsigned short*)(ws + 0);
    unsigned short* msg_w2t  = (unsigned short*)(ws + 196608);
    unsigned short* edge_w1t = (unsigned short*)(ws + 262144);
    unsigned short* edge_w2t = (unsigned short*)(ws + 458752);
    unsigned short* node_w1t = (unsigned short*)(ws + 524288);
    unsigned short* node_w2t = (unsigned short*)(ws + 655360);
    unsigned short* Wnt      = (unsigned short*)(ws + 720896);
    unsigned short* Wet      = (unsigned short*)(ws + 753664);
    float*          agg      = (float*)        (ws + 786432);

    float* out_nodes = (float*)d_out;
    float* out_edges = out_nodes + (long)N * DD;

    auto tgrid = [](long n) { return dim3((unsigned)((n + 255) / 256)); };
    transpose_to_bf16<<<tgrid(384L * 256), 256, 0, stream>>>(msg_w1,  msg_w1t,  384, 256);
    transpose_to_bf16<<<tgrid(256L * 128), 256, 0, stream>>>(msg_w2,  msg_w2t,  256, 128);
    transpose_to_bf16<<<tgrid(384L * 256), 256, 0, stream>>>(edge_w1, edge_w1t, 384, 256);
    transpose_to_bf16<<<tgrid(256L * 128), 256, 0, stream>>>(edge_w2, edge_w2t, 256, 128);
    transpose_to_bf16<<<tgrid(256L * 256), 256, 0, stream>>>(node_w1, node_w1t, 256, 256);
    transpose_to_bf16<<<tgrid(256L * 128), 256, 0, stream>>>(node_w2, node_w2t, 256, 128);
    transpose_to_bf16<<<tgrid(128L * 128), 256, 0, stream>>>(Wn,      Wnt,      128, 128);
    transpose_to_bf16<<<tgrid(128L * 128), 256, 0, stream>>>(We,      Wet,      128, 128);

    zero_f32<<<tgrid((long)N * DD), 256, 0, stream>>>(agg, (long)N * DD);

    const size_t lds_edge = (size_t)(ET * ASTR + ET * NSTR) * sizeof(unsigned short);
    const size_t lds_node = (size_t)(2 * ET * NSTR) * sizeof(unsigned short);
    const dim3 gE((E + ET - 1) / ET), gN((N + ET - 1) / ET), blk(256);

    // message path: builds message_in, MLP, atomic segment-sum into agg
    edge_mlp_kernel<0><<<gE, blk, lds_edge, stream>>>(
        nodes, edges, senders, receivers,
        msg_w1t, msg_b1, msg_w2t, msg_b2, Wet, ln_s, ln_b,
        agg, out_edges, E);

    // node path (depends on agg; same stream => ordered)
    node_mlp_kernel<<<gN, blk, lds_node, stream>>>(
        nodes, agg, node_w1t, node_b1, node_w2t, node_b2, Wnt,
        ln_s, ln_b, out_nodes, N);

    // edge path: independent of agg
    edge_mlp_kernel<1><<<gE, blk, lds_edge, stream>>>(
        nodes, edges, senders, receivers,
        edge_w1t, edge_b1, edge_w2t, edge_b2, Wet, ln_s, ln_b,
        agg, out_edges, E);
}
